// AdaptiveGridSelfAttention_13778255086191
// MI455X (gfx1250) — matrix operand order repaired and checked by reading, not compiled, not run on hardware
//
#include <hip/hip_runtime.h>
#include <hip/hip_bf16.h>

typedef _Float16 half16 __attribute__((ext_vector_type(16)));
typedef _Float16 half8v __attribute__((ext_vector_type(8)));
typedef float    float8 __attribute__((ext_vector_type(8)));
typedef float    float4v __attribute__((ext_vector_type(4)));

#define LDH  72    // half stride: 144 B = 16*9 -> every fragment row 16B aligned
#define LDHH 264   // half stride for 256-wide h: 528 B = 16*33

__device__ __forceinline__ float8 wmma_f16(const half16& a, const half16& b, const float8& c) {
    // D(16x16 f32) = A(16x32 f16) * B(32x16 f16) + C
    return __builtin_amdgcn_wmma_f32_16x16x32_f16(false, a, false, b, (short)0, c, false, false);
}

// A fragment (16x32) from half LDS, row-major [m][k], stride lda (16B-aligned rows).
// lane: m = m0 + (lane&15); e<8 -> K = kk + (lane>>4)*8 + e; e>=8 -> +16.  2x ds_load_b128.
__device__ __forceinline__ half16 loadA_lds(const _Float16* A, int lda, int m0, int kk,
                                            int l15, int lh) {
    const _Float16* row = A + (m0 + l15) * lda + kk + lh * 8;
    half8v lo = *(const half8v*)(row);
    half8v hi = *(const half8v*)(row + 16);
    return __builtin_shufflevector(lo, hi, 0,1,2,3,4,5,6,7,8,9,10,11,12,13,14,15);
}

// B fragment (32x16) from TRANSPOSED half LDS storage BT[n][k], stride ldb.
// lane: n = n0 + (lane&15); e<8 -> K = kk + (lane>>4)*8 + e; e>=8 -> +16.  2x ds_load_b128.
__device__ __forceinline__ half16 loadBT_lds(const _Float16* BT, int ldb, int kk, int n0,
                                             int l15, int lh) {
    const _Float16* row = BT + (n0 + l15) * ldb + kk + lh * 8;
    half8v lo = *(const half8v*)(row);
    half8v hi = *(const half8v*)(row + 16);
    return __builtin_shufflevector(lo, hi, 0,1,2,3,4,5,6,7,8,9,10,11,12,13,14,15);
}

// A fragment from global row-major f32 weights (hot in L2/L0): 4x global_load_b128 + cvt.
__device__ __forceinline__ half16 loadA_gf32(const float* A, int lda, int m0, int kk,
                                             int l15, int lh) {
    const float* row = A + (size_t)(m0 + l15) * lda + kk + lh * 8;
    float4v a0 = *(const float4v*)(row);
    float4v a1 = *(const float4v*)(row + 4);
    float4v a2 = *(const float4v*)(row + 16);
    float4v a3 = *(const float4v*)(row + 20);
    half16 a;
#pragma unroll
    for (int e = 0; e < 4; ++e) {
        a[e]      = (_Float16)a0[e];
        a[4 + e]  = (_Float16)a1[e];
        a[8 + e]  = (_Float16)a2[e];
        a[12 + e] = (_Float16)a3[e];
    }
    return a;
}

// ---------------- Kernel 1: QKV + grid self-attention + residual -> y ----------------
// One block = one (batch, 8x8 tile) group: 64 channels x 64 positions. 8 wave32.
__global__ __launch_bounds__(256)
void attn_kernel(const float* __restrict__ x,
                 const float* __restrict__ wq, const float* __restrict__ bq,
                 const float* __restrict__ wk, const float* __restrict__ bk,
                 const float* __restrict__ wv, const float* __restrict__ bv,
                 float* __restrict__ y) {
    __shared__ __align__(16) unsigned char smem[46080];
    _Float16* xT    = (_Float16*)(smem);          // [64 pos][72 c]   Bt of x
    _Float16* wlds  = (_Float16*)(smem +  9216);  // [64][72]         staged weight (A)
    _Float16* qT    = (_Float16*)(smem + 18432);  // [64 kpos][72 c]  A of attn mm
    _Float16* kT    = (_Float16*)(smem + 27648);  // [64 lpos][72 c]  Bt of k
    _Float16* vh    = (_Float16*)(smem + 36864);  // [64 c][72 l]     A of out mm
    float*    attnFT= (float*)(smem);             // [64 l][72 kpos]  f32, overlays xT+wlds (18432B)
    _Float16* attnP = qT;                         // [64 kpos][72 l]  probs f16, overlays dead qT

    const int tid  = threadIdx.x;
    const int lane = tid & 31, wave = tid >> 5;
    const int l15  = lane & 15, lh = lane >> 4;

    const int blk = blockIdx.x;
    const int b   = blk >> 10;                    // 1024 tiles per batch (32x32 grid)
    const int t   = blk & 1023;
    const int y0  = (t >> 5) << 3;
    const int x0  = (t & 31) << 3;

    const float* xbase = x + (size_t)b * 64 * 65536;

    // stage x tile -> xT[pos][c] (f16); global reads are 16B vector loads
#pragma unroll
    for (int i = 0; i < 4; ++i) {
        int vi = tid + i * 256;                   // [0,1024)
        int c = vi >> 4;
        int p = (vi & 15) * 4;                    // 4 positions, same 8-px row
        float4v v = *(const float4v*)(xbase + (size_t)c * 65536
                                      + (size_t)(y0 + (p >> 3)) * 256 + x0 + (p & 7));
#pragma unroll
        for (int q2 = 0; q2 < 4; ++q2) xT[(p + q2) * LDH + c] = (_Float16)v[q2];
    }
    __syncthreads();

    // stage a 64x64 f32 weight -> wlds (f16, row-major, packed half8 stores)
    auto stage_w = [&](const float* W) {
#pragma unroll
        for (int i = 0; i < 2; ++i) {
            int vi = tid + i * 256;               // [0,512)
            int m = vi >> 3, k8 = (vi & 7) * 8;
            const float* src = W + m * 64 + k8;
            float4v a = *(const float4v*)(src);
            float4v c = *(const float4v*)(src + 4);
            half8v h = { (_Float16)a[0], (_Float16)a[1], (_Float16)a[2], (_Float16)a[3],
                         (_Float16)c[0], (_Float16)c[1], (_Float16)c[2], (_Float16)c[3] };
            *(half8v*)(wlds + m * LDH + k8) = h;
        }
    };

    // 64x64 output = 4x4 tiles of 16x16; wave w: m-tile (w&3), n-tiles (w>>2)*2..+1
    auto mm_qkv = [&](const float* W, const float* bias, _Float16* dst, bool toBT) {
        stage_w(W);
        __syncthreads();
        int m0 = (wave & 3) * 16;
#pragma unroll
        for (int j = 0; j < 2; ++j) {
            int n0 = (wave >> 2) * 32 + j * 16;
            float8 acc = {};
#pragma unroll
            for (int kk = 0; kk < 64; kk += 32) {
                half16 a  = loadA_lds(wlds, LDH, m0, kk, l15, lh);
                half16 bf = loadBT_lds(xT,  LDH, kk, n0, l15, lh);
                acc = wmma_f16(a, bf, acc);
            }
            int n = n0 + l15;
            if (toBT) {      // store transposed [n][m]: 8 contiguous halves -> 1x ds_store_b128
                half8v h;
#pragma unroll
                for (int r = 0; r < 8; ++r) h[r] = (_Float16)(acc[r] + bias[m0 + r + 8 * lh]);
                *(half8v*)(dst + n * LDH + m0 + 8 * lh) = h;
            } else {         // row-major [m][n] scatter (v only; it is the A operand later)
#pragma unroll
                for (int r = 0; r < 8; ++r) {
                    int m = m0 + r + 8 * lh;
                    dst[m * LDH + n] = (_Float16)(acc[r] + bias[m]);
                }
            }
        }
        __syncthreads();
    };

    mm_qkv(wq, bq, qT, true);    // qT[pos][c]
    mm_qkv(wk, bk, kT, true);    // kT[pos][c]
    mm_qkv(wv, bv, vh, false);   // vh[c][pos]

    // attn[kpos][lpos] = sum_c qT[kpos][c] * k[c][lpos]; store transposed f32 (vector stores)
    {
        int m0 = (wave & 3) * 16;
#pragma unroll
        for (int j = 0; j < 2; ++j) {
            int n0 = (wave >> 2) * 32 + j * 16;
            float8 acc = {};
#pragma unroll
            for (int kk = 0; kk < 64; kk += 32) {
                half16 a  = loadA_lds(qT, LDH, m0, kk, l15, lh);
                half16 bf = loadBT_lds(kT, LDH, kk, n0, l15, lh);
                acc = wmma_f16(a, bf, acc);
            }
            int n = n0 + l15;                      // lpos
            float* dstp = attnFT + n * LDH + m0 + 8 * lh;  // m (kpos) contiguous
            float4v lo = { 0.125f * acc[0], 0.125f * acc[1], 0.125f * acc[2], 0.125f * acc[3] };
            float4v hi = { 0.125f * acc[4], 0.125f * acc[5], 0.125f * acc[6], 0.125f * acc[7] };
            *(float4v*)(dstp)     = lo;
            *(float4v*)(dstp + 4) = hi;
        }
    }
    __syncthreads();

    // row softmax (over l); attnFT is [l][kpos] so a row is a conflict-free column walk.
    // write probs row-major attnP[kpos][l] == Bt storage for the output matmul.
    if (tid < 64) {
        const int row = tid;                       // kpos
        float mx = -1e30f;
        for (int l = 0; l < 64; ++l) mx = fmaxf(mx, attnFT[l * LDH + row]);
        float s = 0.f;
        for (int l = 0; l < 64; ++l) {
            float e = __expf(attnFT[l * LDH + row] - mx);
            attnFT[l * LDH + row] = e;
            s += e;
        }
        float inv = 1.f / s;
        for (int l = 0; l < 64; ++l)
            attnP[row * LDH + l] = (_Float16)(attnFT[l * LDH + row] * inv);
    }
    __syncthreads();

    // out[c][kpos] = sum_l v[c][l] * attn[kpos][l]; residual f32, store y
    {
        int m0 = (wave & 3) * 16;
#pragma unroll
        for (int j = 0; j < 2; ++j) {
            int n0 = (wave >> 2) * 32 + j * 16;
            float8 acc = {};
#pragma unroll
            for (int kk = 0; kk < 64; kk += 32) {
                half16 a  = loadA_lds(vh,    LDH, m0, kk, l15, lh);
                half16 bf = loadBT_lds(attnP, LDH, kk, n0, l15, lh);
                acc = wmma_f16(a, bf, acc);
            }
            int n = n0 + l15;                      // position
#pragma unroll
            for (int r = 0; r < 8; ++r) {
                int m = m0 + r + 8 * lh;           // channel
                size_t gi = (size_t)(b * 64 + m) * 65536
                          + (size_t)(y0 + (n >> 3)) * 256 + x0 + (n & 7);
                y[gi] = acc[r] + x[gi];
            }
        }
    }
}

// ---------------- Kernel 2: FFN (64 -> 256 gelu -> 64) + residual, in-place on y ----------------
// One block = 64 consecutive pixels of one batch image, all channels.
__global__ __launch_bounds__(256)
void ffn_kernel(const float* __restrict__ w1, const float* __restrict__ b1,
                const float* __restrict__ w2, const float* __restrict__ b2,
                float* __restrict__ y) {
    __shared__ __align__(16) unsigned char smem[43008];
    _Float16* yT = (_Float16*)(smem);          // [64 pix][72 c]    Bt of y
    _Float16* hT = (_Float16*)(smem + 9216);   // [64 pix][264 hid] Bt of gelu(h)

    const int tid  = threadIdx.x;
    const int lane = tid & 31, wave = tid >> 5;
    const int l15  = lane & 15, lh = lane >> 4;

    const int blk = blockIdx.x;
    const int b   = blk >> 10;                  // 1024 pixel-groups per batch
    const size_t p0 = (size_t)(blk & 1023) << 6;
    float* ybase = y + (size_t)b * 64 * 65536 + p0;

    // stage y tile -> yT[pix][c] (f16); 16B vector global reads
#pragma unroll
    for (int i = 0; i < 4; ++i) {
        int vi = tid + i * 256;
        int c = vi >> 4;
        int p = (vi & 15) * 4;
        float4v v = *(const float4v*)(ybase + (size_t)c * 65536 + p);
#pragma unroll
        for (int q2 = 0; q2 < 4; ++q2) yT[(p + q2) * LDH + c] = (_Float16)v[q2];
    }
    __syncthreads();

    // h = gelu(W1 @ y + b1): 256x64 = 16x4 tiles; wave owns m-tiles {2w,2w+1} x all n.
    // store transposed hT[pix][hid] with packed half8 stores.
#pragma unroll
    for (int mi = 0; mi < 2; ++mi) {
        int m0 = (wave * 2 + mi) * 16;
#pragma unroll
        for (int ni = 0; ni < 4; ++ni) {
            int n0 = ni * 16;
            float8 acc = {};
#pragma unroll
            for (int kk = 0; kk < 64; kk += 32) {
                half16 a  = loadA_gf32(w1, 64, m0, kk, l15, lh);
                half16 bf = loadBT_lds(yT, LDH, kk, n0, l15, lh);
                acc = wmma_f16(a, bf, acc);
            }
            int n = n0 + l15;                      // pixel
            half8v h;
#pragma unroll
            for (int r = 0; r < 8; ++r) {
                int m = m0 + r + 8 * lh;           // hidden unit
                float v = acc[r] + b1[m];
                float g = 0.5f * v * (1.f + tanhf(0.7978845608f * (v + 0.044715f * v * v * v)));
                h[r] = (_Float16)g;
            }
            *(half8v*)(hT + n * LDHH + m0 + 8 * lh) = h;
        }
    }
    __syncthreads();

    // out = y + W2 @ h + b2: 64x64; wave: m-tile (w&3), n-tiles (w>>2)*2..+1; K=256
    {
        int m0 = (wave & 3) * 16;
#pragma unroll
        for (int j = 0; j < 2; ++j) {
            int n0 = (wave >> 2) * 32 + j * 16;
            float8 acc = {};
#pragma unroll
            for (int kk = 0; kk < 256; kk += 32) {
                half16 a  = loadA_gf32(w2, 256, m0, kk, l15, lh);
                half16 bf = loadBT_lds(hT, LDHH, kk, n0, l15, lh);
                acc = wmma_f16(a, bf, acc);
            }
            int n = n0 + l15;                      // pixel
#pragma unroll
            for (int r = 0; r < 8; ++r) {
                int m = m0 + r + 8 * lh;           // channel
                size_t gi = (size_t)m * 65536 + n;
                float res = ybase[gi];             // f32 residual, same-lane read-then-write
                ybase[gi] = acc[r] + b2[m] + res;
            }
        }
    }
}

extern "C" void kernel_launch(void* const* d_in, const int* in_sizes, int n_in,
                              void* d_out, int out_size, void* d_ws, size_t ws_size,
                              hipStream_t stream) {
    (void)in_sizes; (void)n_in; (void)out_size; (void)d_ws; (void)ws_size;
    const float* x  = (const float*)d_in[0];
    const float* wq = (const float*)d_in[1];
    const float* bq = (const float*)d_in[2];
    const float* wk = (const float*)d_in[3];
    const float* bk = (const float*)d_in[4];
    const float* wv = (const float*)d_in[5];
    const float* bv = (const float*)d_in[6];
    const float* w1 = (const float*)d_in[7];
    const float* b1 = (const float*)d_in[8];
    const float* w2 = (const float*)d_in[9];
    const float* b2 = (const float*)d_in[10];
    float* out = (float*)d_out;

    // 8 batches * 32*32 grid tiles = 8192 blocks; 256 threads = 8 wave32
    attn_kernel<<<8192, 256, 0, stream>>>(x, wq, bq, wk, bk, wv, bv, out);
    // 8 batches * 1024 pixel-groups of 64 = 8192 blocks; in-place FFN on y
    ffn_kernel<<<8192, 256, 0, stream>>>(w1, b1, w2, b2, out);
}
